// SAE_87445534146954
// MI455X (gfx1250) — matrix operand order repaired and checked by reading, not compiled, not run on hardware
//
#include <hip/hip_runtime.h>
#include <hip/hip_bf16.h>

// ---------------- problem constants ----------------
#define D_IN   768
#define D_SAE  24576
#define K_TOP  64
#define AUX_K  256
#define DEAD_T 200
#define AUX_C  0.03125f
#define BATCH  2048

typedef unsigned short u16;
typedef __attribute__((ext_vector_type(16))) __bf16 v16bf;
typedef __attribute__((ext_vector_type(8)))  float  v8f;

__device__ __forceinline__ u16 f32_to_bf16(float f) {
  unsigned u = __float_as_uint(f);
  unsigned r = u + 0x7FFFu + ((u >> 16) & 1u);   // round-to-nearest-even
  return (u16)(r >> 16);
}
__device__ __forceinline__ unsigned fkey(float f) {  // monotonic float->uint
  unsigned u = __float_as_uint(f);
  return (u & 0x80000000u) ? ~u : (u | 0x80000000u);
}
__device__ __forceinline__ float funkey(unsigned u) {
  unsigned b = (u & 0x80000000u) ? (u & 0x7FFFFFFFu) : ~u;
  return __uint_as_float(b);
}

// ---------------- prep: xb = bf16(x - b_dec) ----------------
__global__ void __launch_bounds__(256)
k_prep_x(const float* __restrict__ x, const float* __restrict__ bdec,
         u16* __restrict__ xb) {
  int i = blockIdx.x * 256 + threadIdx.x;      // BATCH*D_IN threads
  int c = i % D_IN;
  xb[i] = f32_to_bf16(x[i] - bdec[c]);
}

// ---------------- prep: WT[n][k] = bf16(W_enc[k][n]) (LDS tiled transpose) ----
__global__ void __launch_bounds__(256)
k_prep_w(const float* __restrict__ W, u16* __restrict__ WT) {
  __shared__ float tile[32][33];
  int n0 = blockIdx.x * 32, k0 = blockIdx.y * 32;
  int tx = threadIdx.x & 31, ty = threadIdx.x >> 5;   // 32 x 8
  for (int i = ty; i < 32; i += 8)
    tile[i][tx] = W[(size_t)(k0 + i) * D_SAE + n0 + tx];
  __syncthreads();
  for (int i = ty; i < 32; i += 8)
    WT[(size_t)(n0 + i) * D_IN + k0 + tx] = f32_to_bf16(tile[tx][i]);
}

// ---------------- encoder GEMM: pre = xb @ WT^T + b_enc (WMMA bf16) --------
// Each wave: 32(M) x 64(N) tile -> 8 accumulators; each B-fragment feeds 2
// WMMAs, each A-fragment feeds 4 (loads/WMMA = 1.5).
// A 16x32 bf16 layout (ISA 7.12.2): lane l: m=l&15, half=l>>4,
//   elems 0..7  = A[m, k+8*half .. +7], elems 8..15 = A[m, k+16+8*half .. +7].
// B 32x16 bf16: lane l: n=l&15, half=l>>4, elems 0..7 = B[k+8*half .. +7, n],
//   elems 8..15 = B[k+16+8*half .. +7, n] (same k order as A) from WT row n.
// C/D f32 16x16: lane l holds col n=l&15, rows m = 8*(l>>4)+v, v=0..7.
__global__ void __launch_bounds__(256)
k_enc_gemm(const u16* __restrict__ A,   // [BATCH][D_IN] bf16
           const u16* __restrict__ BT,  // [D_SAE][D_IN] bf16
           const float* __restrict__ benc,
           float* __restrict__ pre) {   // [BATCH][D_SAE] f32
  const int lane = threadIdx.x & 31;
  const int wave = threadIdx.x >> 5;
  const int n0   = blockIdx.x * 64;
  const int m0   = blockIdx.y * 256 + wave * 32;   // two 16-row tiles per wave
  const int hlf  = lane >> 4;
  const int l15  = lane & 15;

  const u16* aRow0 = A + (size_t)(m0 + l15) * D_IN + 8 * hlf;       // m-tile 0
  const u16* aRow1 = aRow0 + (size_t)16 * D_IN;                     // m-tile 1
  const u16* bRow0 = BT + (size_t)(n0 + l15) * D_IN + 8 * hlf;      // +t*16*D_IN

  v8f acc[2][4] = {};

#pragma unroll
  for (int k = 0; k < D_IN; k += 32) {
    union { v16bf v; uint4 q[2]; } a0, a1;
    a0.q[0] = *(const uint4*)(aRow0 + k);
    a0.q[1] = *(const uint4*)(aRow0 + k + 16);
    a1.q[0] = *(const uint4*)(aRow1 + k);
    a1.q[1] = *(const uint4*)(aRow1 + k + 16);
#pragma unroll
    for (int t = 0; t < 4; ++t) {
      const u16* br = bRow0 + (size_t)t * 16 * D_IN + k;
      union { v16bf v; uint4 q[2]; } bf;
      bf.q[0] = *(const uint4*)(br);
      bf.q[1] = *(const uint4*)(br + 16);
      acc[0][t] = __builtin_amdgcn_wmma_f32_16x16x32_bf16(
          false, a0.v, false, bf.v, (short)0, acc[0][t], false, false);
      acc[1][t] = __builtin_amdgcn_wmma_f32_16x16x32_bf16(
          false, a1.v, false, bf.v, (short)0, acc[1][t], false, false);
    }
  }

#pragma unroll
  for (int mt = 0; mt < 2; ++mt) {
#pragma unroll
    for (int t = 0; t < 4; ++t) {
      int n = n0 + t * 16 + l15;
      float be = benc[n];
#pragma unroll
      for (int v = 0; v < 8; ++v)
        pre[(size_t)(m0 + 16 * mt + 8 * hlf + v) * D_SAE + n] = acc[mt][t][v] + be;
    }
  }
}

// ---------------- radix top-K per row (11/11/10-bit MSB-first select) ------
template <int KSEL>
__global__ void __launch_bounds__(256)
k_topk(const float* __restrict__ pre, const float* __restrict__ mask,
       int* __restrict__ oidx, float* __restrict__ oval,
       int* __restrict__ flags) {
  __shared__ unsigned bins[2048];
  __shared__ unsigned s_prefix;
  __shared__ int s_need, s_cgt, s_ceq;
  const int row = blockIdx.x, tid = threadIdx.x;
  const float* p = pre + (size_t)row * D_SAE;

  unsigned prefix = 0;
  int need = KSEL;
  const int sh_arr[3] = {21, 10, 0};
  const int nb_arr[3] = {11, 11, 10};
#pragma unroll
  for (int pass = 0; pass < 3; ++pass) {
    const int sh = sh_arr[pass], nb = 1 << nb_arr[pass];
    for (int b = tid; b < 2048; b += 256) bins[b] = 0;
    __syncthreads();
    unsigned hmask = (pass == 0) ? 0u : (0xFFFFFFFFu << (sh + nb_arr[pass]));
    for (int j = tid; j < D_SAE; j += 256) {
      float v = p[j]; if (mask) v *= mask[j];
      unsigned u = fkey(v);
      if ((u & hmask) == prefix) atomicAdd(&bins[(u >> sh) & (nb - 1)], 1u);
    }
    __syncthreads();
    if (tid == 0) {
      int acc = 0, d = nb - 1;
      for (; d >= 0; --d) { acc += (int)bins[d]; if (acc >= need) break; }
      if (d < 0) d = 0;
      s_need = need - (acc - (int)bins[d]);
      s_prefix = prefix | ((unsigned)d << sh);
    }
    __syncthreads();
    prefix = s_prefix; need = s_need;
  }
  const unsigned T = prefix;
  const int base = KSEL - need;           // count strictly greater than T
  if (tid == 0) { s_cgt = 0; s_ceq = 0; }
  __syncthreads();
  for (int j = tid; j < D_SAE; j += 256) {
    float v = p[j]; if (mask) v *= mask[j];
    unsigned u = fkey(v);
    int slot = -1;
    if (u > T)        slot = atomicAdd(&s_cgt, 1);
    else if (u == T) { int e = atomicAdd(&s_ceq, 1); if (e < need) slot = base + e; }
    if (slot >= 0) {
      float f = funkey(u);
      float rv = f > 0.f ? f : 0.f;       // relu
      oidx[(size_t)row * KSEL + slot] = j;
      oval[(size_t)row * KSEL + slot] = rv;
      if (flags && rv > 0.f) flags[j] = 1;
    }
  }
}

// ---------------- dead-neuron stats --------------------------------------
__global__ void __launch_bounds__(256)
k_stats(const int* __restrict__ stats, const int* __restrict__ flags,
        float* __restrict__ dmask, int* __restrict__ numdead) {
  __shared__ int cnt;
  int i = blockIdx.x * 256 + threadIdx.x;
  if (threadIdx.x == 0) cnt = 0;
  __syncthreads();
  int ns = stats[i] * (flags[i] ? 0 : 1) + 1;
  int dead = ns > DEAD_T;
  dmask[i] = dead ? 1.f : 0.f;
  if (dead) atomicAdd(&cnt, 1);
  __syncthreads();
  if (threadIdx.x == 0) atomicAdd(numdead, cnt);
}

// ---------------- sparse decode: out[row] = sum val_i * Wdec[idx_i] + bdec -
__global__ void __launch_bounds__(256)
k_decode(const int* __restrict__ idx, const float* __restrict__ val, int ksel,
         const float* __restrict__ Wdec, const float* __restrict__ bdec,
         float* __restrict__ out) {
  __shared__ int   sidx[256];
  __shared__ float sval[256];
  const int row = blockIdx.x, tid = threadIdx.x;
  float acc[3];
#pragma unroll
  for (int i = 0; i < 3; ++i) acc[i] = bdec[tid + i * 256];
  for (int b = 0; b < ksel; b += 256) {
    int t = b + tid;
    sidx[tid] = (t < ksel) ? idx[(size_t)row * ksel + t] : 0;
    sval[tid] = (t < ksel) ? val[(size_t)row * ksel + t] : 0.f;
    __syncthreads();
    int lim = (ksel - b < 256) ? (ksel - b) : 256;
    for (int q = 0; q < lim; ++q) {
      // hide gather latency: prefetch next W_dec row (global_prefetch_b8)
      if (q + 1 < lim)
        __builtin_prefetch(Wdec + (size_t)sidx[q + 1] * D_IN + tid, 0, 3);
      float v = sval[q];
      if (v != 0.f) {
        const float* w = Wdec + (size_t)sidx[q] * D_IN;
#pragma unroll
        for (int i = 0; i < 3; ++i) acc[i] += v * w[tid + i * 256];
      }
    }
    __syncthreads();
  }
#pragma unroll
  for (int i = 0; i < 3; ++i) out[(size_t)row * D_IN + tid + i * 256] = acc[i];
}

// ---------------- reduction 1: col sums + SSEs -----------------------------
__global__ void __launch_bounds__(256)
k_reduce1(const float* __restrict__ x, const float* __restrict__ rec,
          const float* __restrict__ aux, const float* __restrict__ bdec,
          float* __restrict__ colx, float* __restrict__ colt,
          float* __restrict__ sse /*[0]=main,[1]=aux*/) {
  __shared__ float red[256];
  const int tid = threadIdx.x, r0 = blockIdx.x * 8;
  float cx[3] = {0, 0, 0}, ct[3] = {0, 0, 0}, sm = 0.f, sa = 0.f;
  for (int r = r0; r < r0 + 8; ++r) {
#pragma unroll
    for (int i = 0; i < 3; ++i) {
      int c = tid + i * 256;
      float xv = x[r * D_IN + c], rv = rec[r * D_IN + c];
      float tv = xv - rv + bdec[c];
      float av = aux[r * D_IN + c];
      cx[i] += xv; ct[i] += tv;
      float d = rv - xv; sm += d * d;
      float e = av - tv; sa += e * e;
    }
  }
#pragma unroll
  for (int i = 0; i < 3; ++i) {
    atomicAdd(&colx[tid + i * 256], cx[i]);
    atomicAdd(&colt[tid + i * 256], ct[i]);
  }
  red[tid] = sm; __syncthreads();
  for (int s = 128; s > 0; s >>= 1) { if (tid < s) red[tid] += red[tid + s]; __syncthreads(); }
  if (tid == 0) atomicAdd(&sse[0], red[0]);
  __syncthreads();
  red[tid] = sa; __syncthreads();
  for (int s = 128; s > 0; s >>= 1) { if (tid < s) red[tid] += red[tid + s]; __syncthreads(); }
  if (tid == 0) atomicAdd(&sse[1], red[0]);
}

// ---------------- reduction 2: naive SSEs vs column means ------------------
__global__ void __launch_bounds__(256)
k_reduce2(const float* __restrict__ x, const float* __restrict__ rec,
          const float* __restrict__ bdec, const float* __restrict__ colx,
          const float* __restrict__ colt, float* __restrict__ naive) {
  __shared__ float red[256];
  const int tid = threadIdx.x, r0 = blockIdx.x * 8;
  float mu[3], mt[3], nm = 0.f, na = 0.f;
#pragma unroll
  for (int i = 0; i < 3; ++i) {
    mu[i] = colx[tid + i * 256] * (1.f / BATCH);
    mt[i] = colt[tid + i * 256] * (1.f / BATCH);
  }
  for (int r = r0; r < r0 + 8; ++r) {
#pragma unroll
    for (int i = 0; i < 3; ++i) {
      int c = tid + i * 256;
      float xv = x[r * D_IN + c], rv = rec[r * D_IN + c];
      float tv = xv - rv + bdec[c];
      float d = mu[i] - xv; nm += d * d;
      float e = mt[i] - tv; na += e * e;
    }
  }
  red[tid] = nm; __syncthreads();
  for (int s = 128; s > 0; s >>= 1) { if (tid < s) red[tid] += red[tid + s]; __syncthreads(); }
  if (tid == 0) atomicAdd(&naive[0], red[0]);
  __syncthreads();
  red[tid] = na; __syncthreads();
  for (int s = 128; s > 0; s >>= 1) { if (tid < s) red[tid] += red[tid + s]; __syncthreads(); }
  if (tid == 0) atomicAdd(&naive[1], red[0]);
}

// ---------------- finalize scalars ----------------------------------------
__global__ void k_finalize(const float* __restrict__ scal, float* __restrict__ tail) {
  // scal: [0]=sse_main [1]=sse_aux [2]=naive_main [3]=naive_aux [4]=numdead(int)
  float mse = scal[0] / scal[2];
  float r = scal[1] / scal[3];
  if (!(r == r)) r = 0.f;                                    // nan -> 0
  else if (r >  3.402823466e38f) r =  3.402823466e38f;       // +inf clamp
  else if (r < -3.402823466e38f) r = -3.402823466e38f;
  int nd = ((const int*)scal)[4];
  float auxl = (nd > 0) ? AUX_C * r : 0.f;
  tail[0] = mse + auxl;   // loss
  tail[1] = mse;          // mse_loss
  tail[2] = auxl;         // aux_loss
  tail[3] = (float)nd;    // num_dead
}

// ---------------- launcher -------------------------------------------------
extern "C" void kernel_launch(void* const* d_in, const int* in_sizes, int n_in,
                              void* d_out, int out_size, void* d_ws, size_t ws_size,
                              hipStream_t stream) {
  const float* x    = (const float*)d_in[0];   // [2048,768]
  const float* Wenc = (const float*)d_in[1];   // [768,24576]
  const float* Wdec = (const float*)d_in[2];   // [24576,768]
  const float* benc = (const float*)d_in[3];   // [24576]
  const float* bdec = (const float*)d_in[4];   // [768]
  const int*   st   = (const int*)d_in[5];     // [24576]
  float* out = (float*)d_out;                  // recons | loss,mse,aux,ndead

  char* w = (char*)d_ws;
  size_t off = 0;
  auto carve = [&](size_t bytes) { size_t o = off; off += (bytes + 255) & ~(size_t)255; return o; };
  size_t o_pre  = carve((size_t)BATCH * D_SAE * 4);
  size_t o_xb   = carve((size_t)BATCH * D_IN * 2);
  size_t o_wT   = carve((size_t)D_SAE * D_IN * 2);
  size_t o_tidx = carve((size_t)BATCH * K_TOP * 4);
  size_t o_tval = carve((size_t)BATCH * K_TOP * 4);
  size_t o_aidx = carve((size_t)BATCH * AUX_K * 4);
  size_t o_aval = carve((size_t)BATCH * AUX_K * 4);
  size_t o_flag = carve((size_t)D_SAE * 4);
  size_t o_dmsk = carve((size_t)D_SAE * 4);
  size_t o_aux  = carve((size_t)BATCH * D_IN * 4);
  size_t o_colx = carve((size_t)D_IN * 4);
  size_t o_colt = carve((size_t)D_IN * 4);
  size_t o_scal = carve(64);

  float* pre   = (float*)(w + o_pre);
  u16*   xb    = (u16*)(w + o_xb);
  u16*   wT    = (u16*)(w + o_wT);
  int*   tidx  = (int*)(w + o_tidx);
  float* tval  = (float*)(w + o_tval);
  int*   aidx  = (int*)(w + o_aidx);
  float* aval  = (float*)(w + o_aval);
  int*   flags = (int*)(w + o_flag);
  float* dmsk  = (float*)(w + o_dmsk);
  float* auxr  = (float*)(w + o_aux);
  float* colx  = (float*)(w + o_colx);
  float* colt  = (float*)(w + o_colt);
  float* scal  = (float*)(w + o_scal);
  int*   ndead = (int*)(scal + 4);

  // zero accumulators (flags; colx..scal contiguous)
  hipMemsetAsync(w + o_flag, 0, (size_t)D_SAE * 4, stream);
  hipMemsetAsync(w + o_colx, 0, o_scal - o_colx + 64, stream);

  k_prep_x<<<(BATCH * D_IN) / 256, 256, 0, stream>>>(x, bdec, xb);
  k_prep_w<<<dim3(D_SAE / 32, D_IN / 32), 256, 0, stream>>>(Wenc, wT);
  k_enc_gemm<<<dim3(D_SAE / 64, BATCH / 256), 256, 0, stream>>>(xb, wT, benc, pre);

  k_topk<K_TOP><<<BATCH, 256, 0, stream>>>(pre, nullptr, tidx, tval, flags);
  k_stats<<<D_SAE / 256, 256, 0, stream>>>(st, flags, dmsk, ndead);
  k_topk<AUX_K><<<BATCH, 256, 0, stream>>>(pre, dmsk, aidx, aval, nullptr);

  k_decode<<<BATCH, 256, 0, stream>>>(tidx, tval, K_TOP, Wdec, bdec, out);
  k_decode<<<BATCH, 256, 0, stream>>>(aidx, aval, AUX_K, Wdec, bdec, auxr);

  k_reduce1<<<BATCH / 8, 256, 0, stream>>>(x, out, auxr, bdec, colx, colt, scal);
  k_reduce2<<<BATCH / 8, 256, 0, stream>>>(x, out, bdec, colx, colt, scal + 2);
  k_finalize<<<1, 1, 0, stream>>>(scal, out + (size_t)BATCH * D_IN);
}
